// FullCausalAttention_9543417331818
// MI455X (gfx1250) — matrix operand order repaired and checked by reading, not compiled, not run on hardware
//
#include <hip/hip_runtime.h>
#include <hip/hip_bf16.h>
#include <math.h>

typedef __attribute__((ext_vector_type(16))) _Float16 v16h;
typedef __attribute__((ext_vector_type(8)))  _Float16 v8h;
typedef __attribute__((ext_vector_type(8)))  float    v8f;
typedef __attribute__((ext_vector_type(4)))  int      i32x4;

#define NB  2
#define NT  4096
#define NC  512
#define NH  8
#define HD  64
#define C3  1536
#define PAD 72   // LDS row stride in halves (144B, 16B-aligned, bank-skewed)

#define AS1 __attribute__((address_space(1)))
#define AS3 __attribute__((address_space(3)))

// ---------------- CDNA5 feature probes ----------------
#if __has_builtin(__builtin_amdgcn_global_load_async_to_lds_b128)
#define HAVE_ASYNC 1
#else
#define HAVE_ASYNC 0
#endif

#if HAVE_ASYNC
#if __has_builtin(__builtin_amdgcn_s_wait_asynccnt)
#define ASYNC_WAIT(n) __builtin_amdgcn_s_wait_asynccnt(n)
#else
#define ASYNC_WAIT(n) asm volatile("s_wait_asynccnt " #n ::: "memory")
#endif
#else
#define ASYNC_WAIT(n) ((void)0)
#endif

#if __has_builtin(__builtin_amdgcn_ds_swizzle)
// group-of-32 swizzle: and=0x1f, or=0, xor=m  -> butterfly within 16-lane halves
#define SWIZ_XOR_I(x, m) __builtin_amdgcn_ds_swizzle((x), (((m) << 10) | 0x1f))
#define XOR16F(x, m) __int_as_float(SWIZ_XOR_I(__float_as_int(x), m))
#else
#define XOR16F(x, m) __shfl_xor((x), (m), 16)
#endif

static __device__ __forceinline__ v8h ld8(const _Float16* p){
  return *reinterpret_cast<const v8h*>(p);
}
static __device__ __forceinline__ v16h cat16(v8h lo, v8h hi){
  return __builtin_shufflevector(lo, hi, 0,1,2,3,4,5,6,7,8,9,10,11,12,13,14,15);
}
static __device__ __forceinline__ v8f wmma16(v16h a, v16h b, v8f c){
  return __builtin_amdgcn_wmma_f32_16x16x32_f16(false, a, false, b, (short)0, c, false, false);
}

#if HAVE_ASYNC
static __device__ __forceinline__ void cp_async16(_Float16* lds, const _Float16* g){
  __builtin_amdgcn_global_load_async_to_lds_b128(
      (AS1 i32x4*)(g), (AS3 i32x4*)(lds), 0, 0);
}
#endif

// stage one 64x64 f16 tile (row-major, PAD-strided) into LDS; 128 threads
static __device__ __forceinline__ void stage_tile(const _Float16* gbase,
                                                  _Float16* ldsbase, int tid){
#pragma unroll
  for (int i = 0; i < 4; ++i){
    int idx = tid + i * 128;
    int r = idx >> 3, c = idx & 7;             // 64 rows x 8 chunks of 8 halves
    const _Float16* g = gbase + (size_t)r * C3 + c * 8;
    _Float16* l = ldsbase + r * PAD + c * 8;
#if HAVE_ASYNC
    cp_async16(l, g);
#else
    *reinterpret_cast<v8h*>(l) = ld8(g);
#endif
  }
}

// ---------------- prep kernels ----------------
__global__ void cvt_f32_to_f16(const float* __restrict__ x, _Float16* __restrict__ y, int n){
  int i = blockIdx.x * 256 + threadIdx.x;
  if (i < n) y[i] = (_Float16)x[i];
}

__global__ void transpose_cvt(const float* __restrict__ w, _Float16* __restrict__ wt, int K, int N){
  int i = blockIdx.x * 256 + threadIdx.x;
  if (i < K * N){
    int n = i % N, k = i / N;
    wt[(size_t)n * K + k] = (_Float16)w[i];
  }
}

// ---------------- tiled WMMA GEMM: Y[M,N] = A[M,K] * Wt[N,K]^T ----------------
template<bool F32OUT>
__global__ __launch_bounds__(128) void gemm_f16(const _Float16* __restrict__ A,
                                                const _Float16* __restrict__ Wt,
                                                void* __restrict__ out,
                                                int M, int N, int K){
  const int lane = threadIdx.x & 31;
  const int wave = threadIdx.x >> 5;
  const int g    = lane >> 4;
  const int hl   = lane & 15;
  const int rbase = blockIdx.x * 64 + wave * 16;
  const int cbase = blockIdx.y * 64;

  v8f acc[4] = {};
  for (int k = 0; k < K; k += 32){
    const _Float16* pa = A + (size_t)(rbase + hl) * K + k + g * 8;
    v16h a = cat16(ld8(pa), ld8(pa + 16));
#pragma unroll
    for (int n = 0; n < 4; ++n){
      const _Float16* pb = Wt + (size_t)(cbase + n * 16 + hl) * K + k + g * 8;
      v16h b = cat16(ld8(pb), ld8(pb + 16));
      acc[n] = wmma16(a, b, acc[n]);
    }
  }
#pragma unroll
  for (int n = 0; n < 4; ++n){
#pragma unroll
    for (int r = 0; r < 8; ++r){
      int row = rbase + r + g * 8;
      int col = cbase + n * 16 + hl;
      if (F32OUT) ((float*)out)[(size_t)row * N + col] = acc[n][r];
      else        ((_Float16*)out)[(size_t)row * N + col] = (_Float16)acc[n][r];
    }
  }
}

// ---------------- flash attention (causal) ----------------
// block = 128 threads (4 waves), 64-query tile; K double-buffered via async-to-LDS
__global__ __launch_bounds__(128) void attn_fwd(const _Float16* __restrict__ qkv,
                                                _Float16* __restrict__ aout){
  __shared__ _Float16 ldsK[2][64 * PAD];   // K tiles, double-buffered (async target)
  __shared__ _Float16 ldsV[64 * PAD];      // V^T : [d][key]
  __shared__ _Float16 ldsP[4 * 16 * PAD];  // per-wave P

  const int tid  = threadIdx.x;
  const int lane = tid & 31;
  const int wave = tid >> 5;
  const int g    = lane >> 4;
  const int hl   = lane & 15;
  const int qt = blockIdx.x;
  const int h  = blockIdx.y;
  const int b  = blockIdx.z;

  const size_t rowbase = (size_t)b * NT;
  const int qoff = h * HD;
  const int koff = NC + h * HD;
  const int voff = 2 * NC + h * HD;

  // Q fragments, pre-scaled by 1/sqrt(64)=0.125 (exact in f16)
  v16h aq[2];
  {
    const size_t qrow = rowbase + (size_t)qt * 64 + wave * 16 + hl;
#pragma unroll
    for (int c = 0; c < 2; ++c){
      const _Float16* p = qkv + qrow * C3 + qoff + c * 32 + g * 8;
      v16h a = cat16(ld8(p), ld8(p + 16));
#pragma unroll
      for (int i = 0; i < 16; ++i) a[i] = a[i] * (_Float16)0.125f;
      aq[c] = a;
    }
  }

  v8f o[4] = {};
  float mrow[8], lrow[8];
#pragma unroll
  for (int r = 0; r < 8; ++r){ mrow[r] = -3.0e38f; lrow[r] = 0.0f; }

  // prologue: stage K tile 0 into buffer 0
  stage_tile(qkv + rowbase * C3 + koff, &ldsK[0][0], tid);

  for (int kt = 0; kt <= qt; ++kt){
    const int buf = kt & 1;
    const size_t krow0 = rowbase + (size_t)kt * 64;

    __syncthreads();   // (A) everyone done reading ldsK[buf^1] / ldsV / ldsP

    // prefetch next K tile into the other buffer while we work on this one
    if (kt < qt)
      stage_tile(qkv + (krow0 + 64) * C3 + koff, &ldsK[buf ^ 1][0], tid);

    // stage V^T: vector global loads, transposed half stores
#pragma unroll
    for (int i = 0; i < 4; ++i){
      int idx = tid + i * 128;
      int key = idx >> 3, c = idx & 7;
      v8h v = ld8(qkv + (krow0 + key) * C3 + voff + c * 8);
#pragma unroll
      for (int j = 0; j < 8; ++j)
        ldsV[(c * 8 + j) * PAD + key] = v[j];
    }

    // wait for THIS tile's async ops only (next tile's 4 are the newest)
    if (kt < qt) { ASYNC_WAIT(4); } else { ASYNC_WAIT(0); }
    __syncthreads();   // (B) K tile + V^T visible block-wide

    // S = Q * K^T from LDS (shared across all 4 waves)
    const _Float16* kb = &ldsK[buf][0];
    v8f s[4];
#pragma unroll
    for (int n = 0; n < 4; ++n){
      v8f acc = {};
#pragma unroll
      for (int c = 0; c < 2; ++c){
        const _Float16* p = kb + (n * 16 + hl) * PAD + c * 32 + g * 8;
        v16h bk = cat16(*(const v8h*)p, *(const v8h*)(p + 16));
        acc = wmma16(aq[c], bk, acc);
      }
      s[n] = acc;
    }

    if (kt == qt){
#pragma unroll
      for (int n = 0; n < 4; ++n){
#pragma unroll
        for (int r = 0; r < 8; ++r){
          int qi = wave * 16 + r + g * 8;
          int ki = n * 16 + hl;
          if (ki > qi) s[n][r] = -3.0e38f;
        }
      }
    }

    // online softmax: row max via ds_swizzle butterflies (16-lane groups)
    float alpha[8];
#pragma unroll
    for (int r = 0; r < 8; ++r){
      float v = fmaxf(fmaxf(s[0][r], s[1][r]), fmaxf(s[2][r], s[3][r]));
      v = fmaxf(v, XOR16F(v, 8));
      v = fmaxf(v, XOR16F(v, 4));
      v = fmaxf(v, XOR16F(v, 2));
      v = fmaxf(v, XOR16F(v, 1));
      float mnew = fmaxf(mrow[r], v);
      alpha[r] = __expf(mrow[r] - mnew);
      mrow[r] = mnew;
    }
#pragma unroll
    for (int n = 0; n < 4; ++n)
#pragma unroll
      for (int r = 0; r < 8; ++r)
        s[n][r] = __expf(s[n][r] - mrow[r]);
#pragma unroll
    for (int r = 0; r < 8; ++r){
      float v = s[0][r] + s[1][r] + s[2][r] + s[3][r];
      v += XOR16F(v, 8);
      v += XOR16F(v, 4);
      v += XOR16F(v, 2);
      v += XOR16F(v, 1);
      lrow[r] = lrow[r] * alpha[r] + v;
    }

#pragma unroll
    for (int n = 0; n < 4; ++n)
#pragma unroll
      for (int r = 0; r < 8; ++r)
        o[n][r] *= alpha[r];

    // P: C-layout -> row-major LDS -> A fragments
    _Float16* pw = ldsP + wave * 16 * PAD;
#pragma unroll
    for (int n = 0; n < 4; ++n)
#pragma unroll
      for (int r = 0; r < 8; ++r)
        pw[(r + g * 8) * PAD + n * 16 + hl] = (_Float16)s[n][r];
    __syncthreads();

    v16h ap[2];
#pragma unroll
    for (int c = 0; c < 2; ++c){
      const _Float16* p = pw + hl * PAD + c * 32 + g * 8;
      ap[c] = cat16(*(const v8h*)p, *(const v8h*)(p + 16));
    }
#pragma unroll
    for (int n = 0; n < 4; ++n){
#pragma unroll
      for (int c = 0; c < 2; ++c){
        const _Float16* p = ldsV + (n * 16 + hl) * PAD + c * 32 + g * 8;
        v16h bv = cat16(*(const v8h*)p, *(const v8h*)(p + 16));
        o[n] = wmma16(ap[c], bv, o[n]);
      }
    }
  }

  // normalize + store f16 attention output [B*T, C]
#pragma unroll
  for (int r = 0; r < 8; ++r) lrow[r] = 1.0f / lrow[r];
  const size_t orow0 = rowbase + (size_t)qt * 64 + wave * 16;
#pragma unroll
  for (int n = 0; n < 4; ++n)
#pragma unroll
    for (int r = 0; r < 8; ++r)
      aout[(orow0 + r + g * 8) * NC + h * HD + n * 16 + hl] = (_Float16)(o[n][r] * lrow[r]);
}

extern "C" void kernel_launch(void* const* d_in, const int* in_sizes, int n_in,
                              void* d_out, int out_size, void* d_ws, size_t ws_size,
                              hipStream_t stream){
  const float* x     = (const float*)d_in[0];   // [2,4096,512]
  const float* w_qkv = (const float*)d_in[1];   // [512,1536]
  const float* w_out = (const float*)d_in[2];   // [512,512]
  float* out = (float*)d_out;                   // [2,4096,512]

  char* ws = (char*)d_ws;
  _Float16* xh    = (_Float16*)(ws);                 //  8192*512  f16
  _Float16* wqkvt = (_Float16*)(ws +  8388608);      //  1536*512  f16
  _Float16* wot   = (_Float16*)(ws +  9961472);      //   512*512  f16
  _Float16* qkvh  = (_Float16*)(ws + 10485760);      //  8192*1536 f16
  _Float16* atth  = (_Float16*)(ws + 35651584);      //  8192*512  f16

  const int M = NB * NT;  // 8192

  cvt_f32_to_f16<<<(M * NC + 255) / 256, 256, 0, stream>>>(x, xh, M * NC);
  transpose_cvt<<<(NC * C3 + 255) / 256, 256, 0, stream>>>(w_qkv, wqkvt, NC, C3);
  transpose_cvt<<<(NC * NC + 255) / 256, 256, 0, stream>>>(w_out, wot, NC, NC);

  dim3 g1(M / 64, C3 / 64);
  gemm_f16<false><<<g1, 128, 0, stream>>>(xh, wqkvt, (void*)qkvh, M, C3, NC);

  dim3 ga(NT / 64, NH, NB);
  attn_fwd<<<ga, 128, 0, stream>>>(qkvh, atth);

  dim3 g2(M / 64, NC / 64);
  gemm_f16<true><<<g2, 128, 0, stream>>>(atth, wot, (void*)out, M, NC, NC);
}
